// AGCRNCell_gatembedding3_80977313399318
// MI455X (gfx1250) — matrix operand order repaired and checked by reading, not compiled, not run on hardware
//
#include <hip/hip_runtime.h>
#include <hip/hip_bf16.h>
#include <math.h>

// ---------------------------------------------------------------------------
// AGCRN cell fused pipeline for MI455X (gfx1250), wave32 + WMMA bf16.
// Main Chebyshev GEMMs: A-frags via global b128 loads, B panel staged into
// LDS by the Tensor Data Mover (double-buffered, s_wait_tensorcnt + barriers),
// B-frags via ds_load_b128 with TDM-generated 16B row padding (no bank
// conflicts). Feature GEMMs: transposed WMMA with pre-padded bf16 weights.
// ---------------------------------------------------------------------------

typedef __bf16 bf16_t;
typedef __attribute__((ext_vector_type(16))) __bf16 v16bf;
typedef __attribute__((ext_vector_type(8)))  float  v8f;
typedef __attribute__((ext_vector_type(4)))  unsigned int v4u;
typedef __attribute__((ext_vector_type(8)))  int v8i;
typedef __attribute__((ext_vector_type(4)))  int v4i;

#define BN     16384   // B*N rows
#define NODES  1024
#define CPAD   160     // padded channel count per block (130 -> 160)
#define CTILES 9       // compute channels 0..143 (144..159 stay zero)
#define KPAD   800     // 5 blocks * 160 for feature GEMM K
#define OG     192     // gate outputs
#define OU     64      // update / mlp outputs
#define KC     128     // k-chunk staged in LDS by TDM
#define NCHUNK (NODES / KC)
#define LDSROW 136     // bf16 elems per LDS row: 128 data + 8 pad (272 B)

static __device__ __forceinline__ unsigned short f2bfbits(float f) {
  union { float f; unsigned u; } c; c.f = f;
  unsigned u = c.u;
  u += 0x7fffu + ((u >> 16) & 1u);   // round-to-nearest-even
  return (unsigned short)(u >> 16);
}
static __device__ __forceinline__ bf16_t bits2bf(unsigned short h) {
  union { unsigned short s; bf16_t b; } o; o.s = h; return o.b;
}
static __device__ __forceinline__ bf16_t f2bf(float f) { return bits2bf(f2bfbits(f)); }
static __device__ __forceinline__ float bf2f(unsigned short h) {
  union { unsigned u; float f; } c; c.u = ((unsigned)h) << 16; return c.f;
}

// A-frag (16x32 bf16): per lane 16 contiguous bf16 at p  (two b128 loads)
static __device__ __forceinline__ v16bf ldfragA(const bf16_t* p) {
  union { v16bf v; uint4 q[2]; } u;
  u.q[0] = *(const uint4*)(p);
  u.q[1] = *(const uint4*)(p + 8);
  return u.v;
}
// B-frag (32x16 bf16): per lane 16 contiguous bf16 at p (column of B, n-major)
static __device__ __forceinline__ v16bf ldfragB(const bf16_t* p) {
  union { v16bf v; uint4 q[2]; } u;
  u.q[0] = *(const uint4*)(p);
  u.q[1] = *(const uint4*)(p + 8);
  return u.v;
}
static __device__ __forceinline__ v8f wmma_bf16(v16bf a, v16bf b, v8f c) {
  return __builtin_amdgcn_wmma_f32_16x16x32_bf16(false, a, false, b, (short)0, c, false, false);
}
static __device__ __forceinline__ float sigmoidf_(float v) { return 1.0f / (1.0f + expf(-v)); }

// ---------------------------------------------------------------------------
// TDM: DMA one B-panel chunk [144 rows x KC cols] (bf16, row stride 1024 elems)
// from global into LDS at lds_off, inserting 16B pad after each 256B row
// (pad_interval = 64 DWORDs, pad_amount = 4 DWORDs) -> 272B LDS row stride.
// ---------------------------------------------------------------------------
static __device__ __forceinline__ void tdm_issue_b_chunk(const bf16_t* gsrc,
                                                         unsigned lds_off) {
  unsigned long long ga = (unsigned long long)(uintptr_t)gsrc;
  v4u g0;
  g0[0] = 1u;                                       // count=1, user descriptor
  g0[1] = lds_off;                                  // lds_addr (bytes)
  g0[2] = (unsigned)(ga & 0xffffffffu);             // global_addr[31:0]
  g0[3] = (unsigned)((ga >> 32) & 0x01ffffffu)      // global_addr[56:32]
        | (2u << 30);                               // type = 2 ("image")
  v8i g1;
  g1[0] = (int)((1u << 16)    // data_size: 2 bytes
              | (1u << 20)    // pad_enable
              | (5u << 22)    // pad_interval: every 64 DWORDs (256B = one row)
              | (3u << 25));  // pad_amount: 4 DWORDs (16B)
  g1[1] = (int)(1024u << 16); // tensor_dim0 = 1024 ([63:48] = low16)
  g1[2] = (int)(160u << 16);  // tensor_dim0 hi=0 | tensor_dim1 = 160 ([95:80])
  g1[3] = (int)((unsigned)KC << 16); // tensor_dim1 hi=0 | tile_dim0 = KC
  g1[4] = 144;                // tile_dim1 = 144 rows | tile_dim2 = 0 (2D)
  g1[5] = 1024;               // tensor_dim0_stride low 32 (elements)
  g1[6] = 0;                  // stride0 hi | dim1_stride (unused, 2D tile)
  g1[7] = 0;
  v4i gz = {0, 0, 0, 0};
#if __clang_major__ >= 23
  v8i gz8 = {0, 0, 0, 0, 0, 0, 0, 0};
  __builtin_amdgcn_tensor_load_to_lds(g0, g1, gz, gz, gz8, 0);
#else
  __builtin_amdgcn_tensor_load_to_lds(g0, g1, gz, gz, 0);
#endif
}

// ---------------------------------------------------------------------------
// Prep: transposed/padded bf16 weights.
// ---------------------------------------------------------------------------
__global__ void prep_weights_kernel(const float* __restrict__ Wg,
                                    const float* __restrict__ Wu,
                                    const float* __restrict__ Wm,
                                    bf16_t* __restrict__ Wtg,
                                    bf16_t* __restrict__ Wtu,
                                    bf16_t* __restrict__ Wtm) {
  int i = blockIdx.x * blockDim.x + threadIdx.x;
  const int base1 = 130, base2 = 260, base3 = 520, base4 = 650;
  if (i < OG * KPAD) {
    int o = i / KPAD, kk = i % KPAD, blk = kk / CPAD, off = kk % CPAD;
    float v = 0.f;
    if (off < 130) {
      if (blk == 0)      v = Wg[off * OG + o] + Wg[(390 + off) * OG + o];
      else if (blk == 1) v = Wg[(base1 + off) * OG + o];
      else if (blk == 2) v = Wg[(base2 + off) * OG + o];
      else if (blk == 3) v = Wg[(base3 + off) * OG + o];
      else               v = Wg[(base4 + off) * OG + o];
    }
    Wtg[(size_t)o * KPAD + kk] = f2bf(v);
    return;
  }
  i -= OG * KPAD;
  if (i < OU * KPAD) {
    int o = i / KPAD, kk = i % KPAD, blk = kk / CPAD, off = kk % CPAD;
    float v = 0.f;
    if (off < 130) {
      if (blk == 0)      v = Wu[off * OU + o] + Wu[(390 + off) * OU + o];
      else if (blk == 1) v = Wu[(base1 + off) * OU + o];
      else if (blk == 2) v = Wu[(base2 + off) * OU + o];
      else if (blk == 3) v = Wu[(base3 + off) * OU + o];
      else               v = Wu[(base4 + off) * OU + o];
    }
    Wtu[(size_t)o * KPAD + kk] = f2bf(v);
    return;
  }
  i -= OU * KPAD;
  if (i < OU * CPAD) {
    int o = i / CPAD, kk = i % CPAD;
    float v = (kk < 146) ? Wm[kk * OU + o] : 0.f;
    Wtm[(size_t)o * CPAD + kk] = f2bf(v);
  }
}

// supports f32 -> bf16, 8 elements/thread (b128 in, b128 out)
__global__ void prep_A_kernel(const float* __restrict__ sup,
                              bf16_t* __restrict__ Abf, long long n8) {
  long long i = (long long)blockIdx.x * blockDim.x + threadIdx.x;
  if (i >= n8) return;
  const float4* p = (const float4*)sup + i * 2;
  float4 a = p[0], b = p[1];
  union { uint4 q; unsigned short h[8]; } pk;
  pk.h[0] = f2bfbits(a.x); pk.h[1] = f2bfbits(a.y);
  pk.h[2] = f2bfbits(a.z); pk.h[3] = f2bfbits(a.w);
  pk.h[4] = f2bfbits(b.x); pk.h[5] = f2bfbits(b.y);
  pk.h[6] = f2bfbits(b.z); pk.h[7] = f2bfbits(b.w);
  ((uint4*)Abf)[i] = pk.q;
}

// Build Fmlp [BN][160], Fg block0 + pads, Fu block0(x)/pads, XtG/XtU channel-major
__global__ void prep_feat_kernel(const float* __restrict__ x,
                                 const float* __restrict__ s1,
                                 const float* __restrict__ s2,
                                 const float* __restrict__ ge,
                                 bf16_t* __restrict__ Fmlp,
                                 bf16_t* __restrict__ Fg,
                                 bf16_t* __restrict__ Fu,
                                 bf16_t* __restrict__ XtG,
                                 bf16_t* __restrict__ XtU) {
  int idx = blockIdx.x * blockDim.x + threadIdx.x;
  if (idx >= BN * CPAD) return;
  int row = idx / CPAD, c = idx % CPAD;
  int b = row >> 10, n = row & 1023;
  float xm;
  if (c < 2)        xm = x[row * 2 + c];
  else if (c < 66)  xm = s1[(size_t)row * 64 + c - 2];
  else if (c < 130) xm = s2[(size_t)row * 64 + c - 66];
  else if (c < 146) xm = ge[(size_t)row * 16 + c - 130];
  else              xm = 0.f;
  Fmlp[(size_t)row * CPAD + c] = f2bf(xm);

  float xg = (c < 130) ? xm : 0.f;
  bf16_t xgb = f2bf(xg);
  Fg[(size_t)row * KPAD + c] = xgb;
  XtG[((size_t)b * CPAD + c) * NODES + n] = xgb;

  bf16_t z = f2bf(0.f);
  if (c < 2) {
    Fu[(size_t)row * KPAD + c] = xgb;
    XtU[((size_t)b * CPAD + c) * NODES + n] = xgb;
  }
  if (c >= 130) {
    Fu[(size_t)row * KPAD + c] = z;
    XtU[((size_t)b * CPAD + c) * NODES + n] = z;
  }
  if (c >= 144) {  // zero pad cols 144..159 of feature blocks 1..4
    #pragma unroll
    for (int blk = 1; blk < 5; ++blk) {
      Fg[(size_t)row * KPAD + blk * CPAD + c] = z;
      Fu[(size_t)row * KPAD + blk * CPAD + c] = z;
    }
  }
}

// ---------------------------------------------------------------------------
// Main Chebyshev pass: per (s,b), D = A @ B over nodes (B = X or Y1).
//  pass1 (XtSub==null): Y1 = A@X  -> Y1t (channel-major) + F block (node-major)
//  pass2 (Yt==null):    Y2 = 2*(A@Y1) - X -> F block (node-major)
// Grid (8,16,2), 256 threads (8 waves); wave = one 16-row m-tile x 9 c-tiles.
// B panel chunks TDM-DMA'd into LDS, double-buffered.
// ---------------------------------------------------------------------------
__global__ __launch_bounds__(256) void agcn_pass_kernel(
    const bf16_t* __restrict__ A,      // [S*B][1024][1024] bf16
    const bf16_t* __restrict__ Bt,     // channel-major [s?][b][160][1024]
    long long bt_s_stride,             // 0 (shared X) or 16*160*1024
    const bf16_t* __restrict__ XtSub,  // [b][160][1024] or null
    bf16_t* __restrict__ Yt,           // [s][b][160][1024] or null
    bf16_t* __restrict__ F,            // [BN][800]
    int f_base, float scale) {
  __shared__ __align__(16) bf16_t ldsB[2][144 * LDSROW];

  const int lane = threadIdx.x & 31;
  const int wave = threadIdx.x >> 5;
  const int lm = lane & 15;
  const int lhi = lane >> 4;
  const int b = blockIdx.y, s = blockIdx.z;
  const int mtile = blockIdx.x * 8 + wave;
  const int m0 = mtile * 16;

  const bf16_t* Ab  = A + ((size_t)(s * 16 + b)) * ((size_t)NODES * NODES);
  const bf16_t* Btb = Bt + (size_t)s * bt_s_stride + (size_t)b * (CPAD * NODES);
  const bf16_t* aLane = Ab + (size_t)(m0 + lm) * NODES + lhi * 16;

  const v8f vzero = {0.f, 0.f, 0.f, 0.f, 0.f, 0.f, 0.f, 0.f};
  v8f acc[CTILES];
  #pragma unroll
  for (int t = 0; t < CTILES; ++t) acc[t] = vzero;

  if (wave == 0)
    tdm_issue_b_chunk(Btb, (unsigned)(uintptr_t)&ldsB[0][0]);

  for (int ch = 0; ch < NCHUNK; ++ch) {
    if (wave == 0) {
      if (ch + 1 < NCHUNK) {
        tdm_issue_b_chunk(Btb + (ch + 1) * KC,
                          (unsigned)(uintptr_t)&ldsB[(ch + 1) & 1][0]);
        __builtin_amdgcn_s_wait_tensorcnt(1);  // chunk ch done, ch+1 in flight
      } else {
        __builtin_amdgcn_s_wait_tensorcnt(0);  // last chunk done
      }
    }
    __syncthreads();  // LDS chunk ch ready for all waves

    const bf16_t* lb = &ldsB[ch & 1][0] + (size_t)lm * LDSROW + lhi * 16;
    const bf16_t* ag = aLane + ch * KC;
    #pragma unroll
    for (int k4 = 0; k4 < KC / 32; ++k4) {
      v16bf af = ldfragA(ag + k4 * 32);
      #pragma unroll
      for (int t = 0; t < CTILES; ++t) {
        v16bf bfr = ldfragB(lb + (size_t)t * 16 * LDSROW + k4 * 32);
        acc[t] = wmma_bf16(af, bfr, acc[t]);
      }
    }
    __syncthreads();  // all waves done with buffer before TDM overwrites it
  }

  const int fcol0 = f_base + s * 320;
  const size_t rowb = (size_t)(b * 1024 + m0 + lhi * 8);
  #pragma unroll
  for (int t = 0; t < CTILES; ++t) {
    const int c = t * 16 + lm;
    float v[8];
    #pragma unroll
    for (int r = 0; r < 8; ++r) v[r] = acc[t][r];
    if (XtSub) {  // pass2: 2*acc - X
      const bf16_t* xp = XtSub + ((size_t)b * CPAD + c) * NODES + m0 + lhi * 8;
      union { uint4 q; unsigned short h[8]; } xv;
      xv.q = *(const uint4*)xp;
      #pragma unroll
      for (int r = 0; r < 8; ++r) v[r] = scale * v[r] - bf2f(xv.h[r]);
    }
    union { uint4 q; unsigned short h[8]; } pk;
    #pragma unroll
    for (int r = 0; r < 8; ++r) pk.h[r] = f2bfbits(v[r]);
    if (Yt) {  // pass1: channel-major copy for pass2's B-frags
      bf16_t* yp = Yt + ((size_t)(s * 16 + b) * CPAD + c) * NODES + m0 + lhi * 8;
      *(uint4*)yp = pk.q;
    }
    bf16_t* fp = F + rowb * KPAD + fcol0 + c;  // node-major feature block
    #pragma unroll
    for (int r = 0; r < 8; ++r) fp[(size_t)r * KPAD] = bits2bf(pk.h[r]);
  }
}

// ---------------------------------------------------------------------------
// Feature GEMMs (transposed: out^T[o,row] = Wt[o,:] . F[row,:]).
// Wave: 1 o-tile x 4 row-tiles. Grid x: 32 blocks * 8 waves = 256 row-groups.
// ---------------------------------------------------------------------------
__global__ __launch_bounds__(256) void gate_gemm_kernel(
    const bf16_t* __restrict__ Wt, const bf16_t* __restrict__ Fg,
    const float* __restrict__ bg, const float* __restrict__ st1,
    const float* __restrict__ st2, bf16_t* __restrict__ Fu,
    bf16_t* __restrict__ XtU, float* __restrict__ rbuf) {
  const int lane = threadIdx.x & 31, wave = threadIdx.x >> 5;
  const int lm = lane & 15, lhi = lane >> 4;
  const int mt = blockIdx.y;             // 0..11  (o range)
  const int ng = blockIdx.x * 8 + wave;  // 0..255 (row group of 64)
  const bf16_t* aLane = Wt + (size_t)(mt * 16 + lm) * KPAD + lhi * 16;
  const bf16_t* bLane = Fg + (size_t)(ng * 64 + lm) * KPAD + lhi * 16;

  const v8f vzero = {0.f, 0.f, 0.f, 0.f, 0.f, 0.f, 0.f, 0.f};
  v8f acc[4];
  #pragma unroll
  for (int t = 0; t < 4; ++t) acc[t] = vzero;
  for (int kb = 0; kb < KPAD; kb += 32) {
    v16bf af = ldfragA(aLane + kb);
    #pragma unroll
    for (int t = 0; t < 4; ++t) {
      v16bf bfr = ldfragB(bLane + (size_t)t * 16 * KPAD + kb);
      acc[t] = wmma_bf16(af, bfr, acc[t]);
    }
  }
  const int obase = mt * 16 + lhi * 8;
  #pragma unroll
  for (int t = 0; t < 4; ++t) {
    int row = ng * 64 + t * 16 + lm;
    int bb = row >> 10, nn = row & 1023;
    if (mt < 8) {  // z1 (o<64) or z2 (64<=o<128) -> candidate features
      const float* sp = (mt < 4 ? st1 : st2) + (size_t)row * 64 + (obase & 63);
      #pragma unroll
      for (int r = 0; r < 8; ++r) {
        float sg = sigmoidf_(acc[t][r] + bg[obase + r]);
        bf16_t cb = f2bf(sg * sp[r]);
        int ch = 2 + obase + r;  // candidate channel
        Fu[(size_t)row * KPAD + ch] = cb;
        XtU[((size_t)bb * CPAD + ch) * NODES + nn] = cb;
      }
    } else {       // r gate (128<=o<192)
      float* rp = rbuf + (size_t)row * 64 + (obase - 128);
      #pragma unroll
      for (int r = 0; r < 8; ++r) rp[r] = sigmoidf_(acc[t][r] + bg[obase + r]);
    }
  }
}

__global__ __launch_bounds__(256) void upd_gemm_kernel(
    const bf16_t* __restrict__ Wt, const bf16_t* __restrict__ Fu,
    const float* __restrict__ bu, const float* __restrict__ rbuf,
    const float* __restrict__ stbuf, float* __restrict__ out) {
  const int lane = threadIdx.x & 31, wave = threadIdx.x >> 5;
  const int lm = lane & 15, lhi = lane >> 4;
  const int mt = blockIdx.y;             // 0..3
  const int ng = blockIdx.x * 8 + wave;  // 0..255
  const bf16_t* aLane = Wt + (size_t)(mt * 16 + lm) * KPAD + lhi * 16;
  const bf16_t* bLane = Fu + (size_t)(ng * 64 + lm) * KPAD + lhi * 16;

  const v8f vzero = {0.f, 0.f, 0.f, 0.f, 0.f, 0.f, 0.f, 0.f};
  v8f acc[4];
  #pragma unroll
  for (int t = 0; t < 4; ++t) acc[t] = vzero;
  for (int kb = 0; kb < KPAD; kb += 32) {
    v16bf af = ldfragA(aLane + kb);
    #pragma unroll
    for (int t = 0; t < 4; ++t) {
      v16bf bfr = ldfragB(bLane + (size_t)t * 16 * KPAD + kb);
      acc[t] = wmma_bf16(af, bfr, acc[t]);
    }
  }
  const int obase = mt * 16 + lhi * 8;
  #pragma unroll
  for (int t = 0; t < 4; ++t) {
    size_t row = (size_t)(ng * 64 + t * 16 + lm);
    #pragma unroll
    for (int r = 0; r < 8; ++r) {
      float hc = tanhf(acc[t][r] + bu[obase + r]);
      float rr = rbuf[row * 64 + obase + r];
      float st = stbuf[row * 64 + obase + r];
      out[row * 64 + obase + r] = rr * st + (1.0f - rr) * hc;
    }
  }
}

__global__ __launch_bounds__(256) void mlp_gemm_kernel(
    const bf16_t* __restrict__ Wt, const bf16_t* __restrict__ Fm,
    const float* __restrict__ bm, const float* __restrict__ st1,
    const float* __restrict__ st2, float* __restrict__ stbuf) {
  const int lane = threadIdx.x & 31, wave = threadIdx.x >> 5;
  const int lm = lane & 15, lhi = lane >> 4;
  const int mt = blockIdx.y;             // 0..3
  const int ng = blockIdx.x * 8 + wave;  // 0..255
  const bf16_t* aLane = Wt + (size_t)(mt * 16 + lm) * CPAD + lhi * 16;
  const bf16_t* bLane = Fm + (size_t)(ng * 64 + lm) * CPAD + lhi * 16;

  const v8f vzero = {0.f, 0.f, 0.f, 0.f, 0.f, 0.f, 0.f, 0.f};
  v8f acc[4];
  #pragma unroll
  for (int t = 0; t < 4; ++t) acc[t] = vzero;
  for (int kb = 0; kb < CPAD; kb += 32) {
    v16bf af = ldfragA(aLane + kb);
    #pragma unroll
    for (int t = 0; t < 4; ++t) {
      v16bf bfr = ldfragB(bLane + (size_t)t * 16 * CPAD + kb);
      acc[t] = wmma_bf16(af, bfr, acc[t]);
    }
  }
  const int obase = mt * 16 + lhi * 8;
  #pragma unroll
  for (int t = 0; t < 4; ++t) {
    size_t row = (size_t)(ng * 64 + t * 16 + lm);
    #pragma unroll
    for (int r = 0; r < 8; ++r) {
      float mr = sigmoidf_(acc[t][r] + bm[obase + r]);
      float a = st1[row * 64 + obase + r];
      float b = st2[row * 64 + obase + r];
      stbuf[row * 64 + obase + r] = mr * a + (1.0f - mr) * b;
    }
  }
}

// ---------------------------------------------------------------------------
extern "C" void kernel_launch(void* const* d_in, const int* in_sizes, int n_in,
                              void* d_out, int out_size, void* d_ws, size_t ws_size,
                              hipStream_t stream) {
  (void)in_sizes; (void)n_in; (void)out_size; (void)ws_size;
  const float* x   = (const float*)d_in[0];
  const float* s1  = (const float*)d_in[1];
  const float* s2  = (const float*)d_in[2];
  const float* ge  = (const float*)d_in[3];
  const float* sup = (const float*)d_in[4];
  const float* Wm  = (const float*)d_in[5];
  const float* bm  = (const float*)d_in[6];
  const float* Wg  = (const float*)d_in[7];
  const float* bg  = (const float*)d_in[8];
  const float* Wu  = (const float*)d_in[9];
  const float* bu  = (const float*)d_in[10];
  float* out = (float*)d_out;

  // workspace carve-up (~148 MB total)
  char* ws = (char*)d_ws;
  size_t off = 0;
  auto carve = [&](size_t bytes) {
    char* p = ws + off;
    off += (bytes + 255) & ~(size_t)255;
    return p;
  };
  bf16_t* Abf  = (bf16_t*)carve((size_t)2 * 16 * 1024 * 1024 * 2); // 67.1 MB
  bf16_t* XtG  = (bf16_t*)carve((size_t)16 * CPAD * NODES * 2);    //  5.2 MB
  bf16_t* XtU  = (bf16_t*)carve((size_t)16 * CPAD * NODES * 2);    //  5.2 MB
  bf16_t* Y1t  = (bf16_t*)carve((size_t)32 * CPAD * NODES * 2);    // 10.5 MB (reused gate->upd)
  bf16_t* Fg   = (bf16_t*)carve((size_t)BN * KPAD * 2);            // 26.2 MB
  bf16_t* Fu   = (bf16_t*)carve((size_t)BN * KPAD * 2);            // 26.2 MB
  bf16_t* Fmlp = (bf16_t*)carve((size_t)BN * CPAD * 2);            //  5.2 MB
  bf16_t* Wtg  = (bf16_t*)carve((size_t)OG * KPAD * 2);
  bf16_t* Wtu  = (bf16_t*)carve((size_t)OU * KPAD * 2);
  bf16_t* Wtm  = (bf16_t*)carve((size_t)OU * CPAD * 2);
  float*  stbuf= (float*)carve((size_t)BN * 64 * 4);               //  4.2 MB
  float*  rbuf = (float*)carve((size_t)BN * 64 * 4);               //  4.2 MB

  { int tot = OG * KPAD + OU * KPAD + OU * CPAD;
    prep_weights_kernel<<<(tot + 255) / 256, 256, 0, stream>>>(Wg, Wu, Wm, Wtg, Wtu, Wtm); }
  { long long n8 = (2ll * 16 * 1024 * 1024) / 8;
    prep_A_kernel<<<(unsigned)((n8 + 255) / 256), 256, 0, stream>>>(sup, Abf, n8); }
  { int tot = BN * CPAD;
    prep_feat_kernel<<<(tot + 255) / 256, 256, 0, stream>>>(x, s1, s2, ge, Fmlp, Fg, Fu, XtG, XtU); }

  // mr MLP -> blended state
  mlp_gemm_kernel<<<dim3(32, 4), 256, 0, stream>>>(Wtm, Fmlp, bm, s1, s2, stbuf);

  const long long y1s = (long long)16 * CPAD * NODES;
  // gate path: Y1 = A@X ; Y2 = 2*A@Y1 - X
  agcn_pass_kernel<<<dim3(8, 16, 2), 256, 0, stream>>>(Abf, XtG, 0, nullptr, Y1t, Fg, 160, 1.0f);
  agcn_pass_kernel<<<dim3(8, 16, 2), 256, 0, stream>>>(Abf, Y1t, y1s, XtG, nullptr, Fg, 320, 2.0f);
  // gate GEMM: sigmoid, build candidate (z1*s1, z2*s2) + r
  gate_gemm_kernel<<<dim3(32, 12), 256, 0, stream>>>(Wtg, Fg, bg, s1, s2, Fu, XtU, rbuf);
  // update path on candidate
  agcn_pass_kernel<<<dim3(8, 16, 2), 256, 0, stream>>>(Abf, XtU, 0, nullptr, Y1t, Fu, 160, 1.0f);
  agcn_pass_kernel<<<dim3(8, 16, 2), 256, 0, stream>>>(Abf, Y1t, y1s, XtU, nullptr, Fu, 320, 2.0f);
  // final: hc = tanh(...), h = r*state + (1-r)*hc
  upd_gemm_kernel<<<dim3(32, 4), 256, 0, stream>>>(Wtu, Fu, bu, rbuf, stbuf, out);
}
